// LHGNN_37907381355076
// MI455X (gfx1250) — matrix operand order repaired and checked by reading, not compiled, not run on hardware
//
#include <hip/hip_runtime.h>
#include <hip/hip_bf16.h>
#include <math.h>

// ---------------------------------------------------------------------------
// LHGNN forward for MI455X (gfx1250, wave32).
// All 1x1 convs + KNN similarity on v_wmma_f32_16x16x32_bf16.
// bf16 GEMM operands are stored NODE-MAJOR ([nodes, channels]) so both A and B
// fragments are pairs of 16B contiguous loads; bf16 outputs are written as one
// v8bf (b128) store per accumulator. The K-loop computes all 4 row-subtiles
// unconditionally (overshoot rows stay inside the weight slab; results only
// stored when in-range), giving a branch-free 10-load / 4-wmma loop body.
// Shapes: M,K % 16 == 0, N % 64 == 0, K % 32 in {0,16} (16-tail = zero group).
// ---------------------------------------------------------------------------

typedef __attribute__((ext_vector_type(16))) __bf16 v16bf;
typedef __attribute__((ext_vector_type(8)))  __bf16 v8bf;
typedef __attribute__((ext_vector_type(8)))  float  v8f;

#define DEV __device__ __forceinline__

static const int BATCH = 32;
static const int K_NEI = 10;

DEV __bf16 f2bf(float f) {
  unsigned u = __builtin_bit_cast(unsigned, f);
  unsigned r = u + 0x7FFFu + ((u >> 16) & 1u);   // round-to-nearest-even
  unsigned short h = (unsigned short)(r >> 16);
  return __builtin_bit_cast(__bf16, h);
}
DEV __bf16 bf_zero() { return __builtin_bit_cast(__bf16, (unsigned short)0); }

DEV float gelu_f(float x) {
  const float c0 = 0.7978845608028654f;          // sqrt(2/pi)
  float t = tanhf(c0 * (x + 0.044715f * x * x * x));
  return 0.5f * x * (1.0f + t);
}

DEV v8bf ld8(const __bf16* p) { return *(const v8bf*)p; }
DEV v16bf pack2(v8bf g0, v8bf g1) {
  v16bf o;
#pragma unroll
  for (int j = 0; j < 8; ++j) { o[j] = g0[j]; o[8 + j] = g1[j]; }
  return o;
}
DEV v8bf zero8() {
  v8bf z;
#pragma unroll
  for (int j = 0; j < 8; ++j) z[j] = bf_zero();
  return z;
}

#define WMMA_BF16(a, b, c) \
  __builtin_amdgcn_wmma_f32_16x16x32_bf16(false, (a), false, (b), (short)0, (c), false, false)

// Fragment layouts (CDNA5 ISA §7.12.2):
//  A (16x32, row-major [M,K]):  lane -> o[j] = A[row, k0+half*8+j], o[8+j] = +16
//  B via node-major BT [N,K]:   lane -> o[j] = BT[col, k0+half*8+j], o[8+j] = +16

// ---------------------------------------------------------------------------
// Batched GEMM: Cf[b] (f32 [M,N]) / CbT[b] (bf16 [N,M]) =
//   act(bn(A * BT[b]^T + bias)) (+ resid[b])
// Grid: x = N/64, y = ceil(M/64), z = batch; 128 threads = 4 waves.
// ---------------------------------------------------------------------------
__global__ void gemm_bf16_wmma_kernel(
    const __bf16* __restrict__ A,                      // [M,K]
    const __bf16* __restrict__ BT, long long b_bs,     // [N,K] per batch
    float* __restrict__ Cf, long long c_bs,            // [M,N] per batch
    __bf16* __restrict__ CbT, long long cb_bs,         // [N,M] per batch
    const float* __restrict__ bias,
    const float* __restrict__ gamma, const float* __restrict__ beta,
    const float* __restrict__ resid,                   // [M,N], batch stride c_bs
    int M, int N, int Kdim, int act) {
  const int lane = threadIdx.x & 31;
  const int wave = threadIdx.x >> 5;
  const int half = lane >> 4;
  const int lr   = lane & 15;
  const int col0 = blockIdx.x * 64 + wave * 16;
  const int rowBase = blockIdx.y * 64;
  const int nt = min(4, (M - rowBase) >> 4);
  if (col0 >= N || nt <= 0) return;                    // wave-uniform
  const int b = blockIdx.z;
  const int col = col0 + lr;

  const __bf16* pb = BT + (long long)b * b_bs + (long long)col * Kdim + half * 8;
  const __bf16* pa = A + (long long)(rowBase + lr) * Kdim + half * 8;
  const long long rS = (long long)16 * Kdim;           // 16 rows of A

  v8f acc0 = {}, acc1 = {}, acc2 = {}, acc3 = {};
  const int Kmain = Kdim & ~31;
  for (int k0 = 0; k0 < Kmain; k0 += 32) {
    v16bf bf = pack2(ld8(pb), ld8(pb + 16));
    v16bf a0 = pack2(ld8(pa),           ld8(pa + 16));
    v16bf a1 = pack2(ld8(pa + rS),      ld8(pa + rS + 16));
    v16bf a2 = pack2(ld8(pa + 2 * rS),  ld8(pa + 2 * rS + 16));
    v16bf a3 = pack2(ld8(pa + 3 * rS),  ld8(pa + 3 * rS + 16));
    acc0 = WMMA_BF16(a0, bf, acc0);
    acc1 = WMMA_BF16(a1, bf, acc1);
    acc2 = WMMA_BF16(a2, bf, acc2);
    acc3 = WMMA_BF16(a3, bf, acc3);
    pa += 32; pb += 32;
  }
  if (Kmain < Kdim) {                                  // 16-wide K tail
    v8bf z = zero8();
    v16bf bf = pack2(ld8(pb), z);
    v16bf a0 = pack2(ld8(pa),          z);
    v16bf a1 = pack2(ld8(pa + rS),     z);
    v16bf a2 = pack2(ld8(pa + 2 * rS), z);
    v16bf a3 = pack2(ld8(pa + 3 * rS), z);
    acc0 = WMMA_BF16(a0, bf, acc0);
    acc1 = WMMA_BF16(a1, bf, acc1);
    acc2 = WMMA_BF16(a2, bf, acc2);
    acc3 = WMMA_BF16(a3, bf, acc3);
  }

  float* Cfb = Cf ? Cf + (long long)b * c_bs : nullptr;
  __bf16* Cbb = CbT ? CbT + (long long)b * cb_bs + (long long)col * M : nullptr;
  const float* Rb = resid ? resid + (long long)b * c_bs : nullptr;
  v8f accs[4] = {acc0, acc1, acc2, acc3};
#pragma unroll
  for (int j = 0; j < 4; ++j) {
    if (j >= nt) break;
    v8f acc = accs[j];
    const int rj = rowBase + j * 16 + half * 8;        // 8 consecutive rows
    v8bf hv;
#pragma unroll
    for (int r2 = 0; r2 < 8; ++r2) {
      int row = rj + r2;
      float v = acc[r2];
      if (bias)  v += bias[row];
      if (gamma) v = v * gamma[row] + beta[row];
      if (act)   v = gelu_f(v);
      if (Rb)    v += Rb[(long long)row * N + col];
      if (Cfb)   Cfb[(long long)row * N + col] = v;
      hv[r2] = f2bf(v);
    }
    if (Cbb) *(v8bf*)(Cbb + rj) = hv;                  // one b128 store
  }
}

// ---------------------------------------------------------------------------
// KNN similarity: D[b,n,m] = 2*dot - sx[n] - sy[m] + rel[n,m]
// xnT [B,N,C] bf16 (queries, node-major), ynT [B,M,C] bf16 (candidates).
// ---------------------------------------------------------------------------
__global__ void dist_bf16_wmma_kernel(
    const __bf16* __restrict__ xnT, const __bf16* __restrict__ ynT,
    const float* __restrict__ sx, const float* __restrict__ sy,
    const float* __restrict__ rel, float* __restrict__ D,
    int N, int Mc, int Cdim) {
  const int lane = threadIdx.x & 31;
  const int wave = threadIdx.x >> 5;
  const int half = lane >> 4;
  const int lr   = lane & 15;
  const int col0 = blockIdx.x * 64 + wave * 16;
  const int rowBase = blockIdx.y * 64;
  if (col0 >= Mc) return;
  const int b = blockIdx.z;
  const int col = col0 + lr;

  const __bf16* pa = xnT + (long long)b * N * Cdim
                   + (long long)(rowBase + lr) * Cdim + half * 8;
  const __bf16* pb = ynT + (long long)b * Mc * Cdim
                   + (long long)col * Cdim + half * 8;
  const long long rS = (long long)16 * Cdim;

  v8f acc0 = {}, acc1 = {}, acc2 = {}, acc3 = {};
  const int Kmain = Cdim & ~31;
  for (int k0 = 0; k0 < Kmain; k0 += 32) {
    v16bf bf = pack2(ld8(pb), ld8(pb + 16));
    v16bf a0 = pack2(ld8(pa),          ld8(pa + 16));
    v16bf a1 = pack2(ld8(pa + rS),     ld8(pa + rS + 16));
    v16bf a2 = pack2(ld8(pa + 2 * rS), ld8(pa + 2 * rS + 16));
    v16bf a3 = pack2(ld8(pa + 3 * rS), ld8(pa + 3 * rS + 16));
    acc0 = WMMA_BF16(a0, bf, acc0);
    acc1 = WMMA_BF16(a1, bf, acc1);
    acc2 = WMMA_BF16(a2, bf, acc2);
    acc3 = WMMA_BF16(a3, bf, acc3);
    pa += 32; pb += 32;
  }
  if (Kmain < Cdim) {
    v8bf z = zero8();
    v16bf bf = pack2(ld8(pb), z);
    v16bf a0 = pack2(ld8(pa),          z);
    v16bf a1 = pack2(ld8(pa + rS),     z);
    v16bf a2 = pack2(ld8(pa + 2 * rS), z);
    v16bf a3 = pack2(ld8(pa + 3 * rS), z);
    acc0 = WMMA_BF16(a0, bf, acc0);
    acc1 = WMMA_BF16(a1, bf, acc1);
    acc2 = WMMA_BF16(a2, bf, acc2);
    acc3 = WMMA_BF16(a3, bf, acc3);
  }

  float* Db = D + (long long)b * N * Mc;
  const float* sxb = sx + (long long)b * N;
  const float syv = sy[(long long)b * Mc + col];
  v8f accs[4] = {acc0, acc1, acc2, acc3};
#pragma unroll
  for (int j = 0; j < 4; ++j) {
    v8f acc = accs[j];
#pragma unroll
    for (int r2 = 0; r2 < 8; ++r2) {
      int row = rowBase + j * 16 + half * 8 + r2;
      Db[(long long)row * Mc + col] =
          2.0f * acc[r2] - sxb[row] - syv + rel[(long long)row * Mc + col];
    }
  }
}

// ----------------------------- scalar kernels ------------------------------

// [B,C,N] f32 -> [B,N,C] bf16 (node-major transpose + convert)
__global__ void cvtT_f32_bf16_kernel(const float* __restrict__ in, __bf16* __restrict__ out,
                                     int C, int N, long long total) {
  long long t = (long long)blockIdx.x * blockDim.x + threadIdx.x;
  if (t >= total) return;
  int n = (int)(t % N); long long r = t / N;
  int c = (int)(r % C); int b = (int)(r / C);
  out[((long long)b * N + n) * C + c] = f2bf(in[t]);
}

// weights: plain convert [M,K] f32 -> bf16
__global__ void cvt_f32_bf16_kernel(const float* __restrict__ in, __bf16* __restrict__ out,
                                    long long n) {
  long long t = (long long)blockIdx.x * blockDim.x + threadIdx.x;
  if (t >= n) return;
  out[t] = f2bf(in[t]);
}

__global__ void transpose_in_kernel(const float* __restrict__ x, float* __restrict__ o,
                                    int T, int F, long long total) {
  long long t = (long long)blockIdx.x * blockDim.x + threadIdx.x;
  if (t >= total) return;
  int tt = (int)(t % T); long long r = t / T;
  int f = (int)(r % F); int b = (int)(r / F);
  o[t] = x[((long long)b * T + tt) * F + f];
}

__global__ void conv3x3_kernel(const float* __restrict__ in, const float* __restrict__ w,
                               const float* __restrict__ bias,
                               const float* __restrict__ gamma, const float* __restrict__ beta,
                               float* __restrict__ out,
                               int Cin, int Cout, int Hin, int Win, int Hout, int Wout,
                               int stride, int ph, int pw, int act, long long total) {
  long long t = (long long)blockIdx.x * blockDim.x + threadIdx.x;
  if (t >= total) return;
  int wo = (int)(t % Wout); long long r = t / Wout;
  int ho = (int)(r % Hout); r /= Hout;
  int co = (int)(r % Cout); int b = (int)(r / Cout);
  const float* inb = in + (long long)b * Cin * Hin * Win;
  float s = 0.0f;
  for (int ci = 0; ci < Cin; ++ci) {
    const float* ip = inb + (long long)ci * Hin * Win;
    const float* wp = w + ((long long)co * Cin + ci) * 9;
#pragma unroll
    for (int kh = 0; kh < 3; ++kh) {
      int hi = ho * stride - ph + kh;
      if (hi < 0 || hi >= Hin) continue;
#pragma unroll
      for (int kw = 0; kw < 3; ++kw) {
        int wi = wo * stride - pw + kw;
        if (wi < 0 || wi >= Win) continue;
        s += ip[(long long)hi * Win + wi] * wp[kh * 3 + kw];
      }
    }
  }
  float v = (s + bias[co]) * gamma[co] + beta[co];
  if (act) v = gelu_f(v);
  out[t] = v;
}

// depthwise 3x3, stride1 pad1; bf16 output node-major [B,N,C]
__global__ void dwconv3x3_kernel(const float* __restrict__ in, const float* __restrict__ w,
                                 const float* __restrict__ bias,
                                 const float* __restrict__ gamma, const float* __restrict__ beta,
                                 __bf16* __restrict__ outT, int C, int H, int W,
                                 long long total) {
  long long t = (long long)blockIdx.x * blockDim.x + threadIdx.x;
  if (t >= total) return;
  int wo = (int)(t % W); long long r = t / W;
  int ho = (int)(r % H); r /= H;
  int c = (int)(r % C); int b = (int)(r / C);
  const float* ip = in + ((t / ((long long)H * W)) * (long long)H * W);
  const float* wp = w + (long long)c * 9;
  float s = 0.0f;
#pragma unroll
  for (int kh = 0; kh < 3; ++kh) {
    int hi = ho - 1 + kh;
    if (hi < 0 || hi >= H) continue;
#pragma unroll
    for (int kw = 0; kw < 3; ++kw) {
      int wi = wo - 1 + kw;
      if (wi < 0 || wi >= W) continue;
      s += ip[(long long)hi * W + wi] * wp[kh * 3 + kw];
    }
  }
  float v = (s + bias[c]) * gamma[c] + beta[c];
  int n = ho * W + wo;
  outT[((long long)b * H * W + n) * C + c] = f2bf(gelu_f(v));
}

__global__ void addpos_kernel(float* __restrict__ actb, const float* __restrict__ pos,
                              long long per, long long total) {
  long long t = (long long)blockIdx.x * blockDim.x + threadIdx.x;
  if (t >= total) return;
  actb[t] += pos[t % per];
}

__global__ void avgpool_kernel(const float* __restrict__ in, float* __restrict__ out,
                               int C, int H, int W, int rr, long long total) {
  long long t = (long long)blockIdx.x * blockDim.x + threadIdx.x;
  if (t >= total) return;
  int Ho = H / rr, Wo = W / rr;
  int wo = (int)(t % Wo); long long r = t / Wo;
  int ho = (int)(r % Ho); r /= Ho;
  int c = (int)(r % C);  int b = (int)(r / C);
  const float* ip = in + ((long long)b * C + c) * H * W;
  float s = 0.0f;
  for (int i = 0; i < rr; ++i)
    for (int j = 0; j < rr; ++j)
      s += ip[(long long)(ho * rr + i) * W + (wo * rr + j)];
  out[t] = s / (float)(rr * rr);
}

// per-node L2 normalize over channels; bf16 node-major output + f32 ||.||^2
__global__ void normalize_kernel(const float* __restrict__ x, __bf16* __restrict__ xnT,
                                 float* __restrict__ sn, int C, int N, long long total) {
  long long t = (long long)blockIdx.x * blockDim.x + threadIdx.x;   // t = b*N + n
  if (t >= total) return;
  int n = (int)(t % N); int b = (int)(t / N);
  const float* xb = x + (long long)b * C * N;
  __bf16* ob = xnT + ((long long)b * N + n) * C;
  float ss = 0.0f;
  for (int c = 0; c < C; ++c) { float v = xb[(long long)c * N + n]; ss += v * v; }
  float inv = 1.0f / (sqrtf(ss) + 1e-12f);
  float s2 = 0.0f;
  for (int c = 0; c < C; ++c) {
    float v = xb[(long long)c * N + n] * inv;
    ob[c] = f2bf(v);                         // contiguous per-thread writes
    s2 += v * v;
  }
  sn[t] = s2;
}

__global__ void topk_kernel(const float* __restrict__ D, int* __restrict__ idxo,
                            int Mc, int kd, int dil, long long rows) {
  long long row = (long long)blockIdx.x * blockDim.x + threadIdx.x;
  if (row >= rows) return;
  const float* d = D + row * Mc;
  float val[32]; int ind[32];
#pragma unroll
  for (int i = 0; i < 32; ++i) { val[i] = -3.4e38f; ind[i] = 0; }
  for (int m = 0; m < Mc; ++m) {
    float v = d[m];
    if (v > val[kd - 1]) {
      int j = kd - 1;
      while (j > 0 && v > val[j - 1]) { val[j] = val[j - 1]; ind[j] = ind[j - 1]; --j; }
      val[j] = v; ind[j] = m;
    }
  }
  int* o = idxo + row * K_NEI;
  for (int i = 0; i < K_NEI; ++i) o[i] = ind[i * dil];
}

// h2T[b,n,2c]=x[b,c,n]; h2T[b,n,2c+1]=max_i(y[b,c,idx[b,n,i]]-x[b,c,n])
__global__ void gather_maxrel_kernel(const float* __restrict__ x, const float* __restrict__ y,
                                     const int* __restrict__ idx, __bf16* __restrict__ outT,
                                     int C, int N, int Mc, long long total) {
  long long t = (long long)blockIdx.x * blockDim.x + threadIdx.x;   // t = ((b*C)+c)*N + n
  if (t >= total) return;
  int n = (int)(t % N); long long r = t / N;
  int c = (int)(r % C); int b = (int)(r / C);
  float xv = x[t];
  const float* yb = y + ((long long)b * C + c) * Mc;
  const int* ib = idx + ((long long)b * N + n) * K_NEI;
  float mx = -3.4e38f;
#pragma unroll
  for (int i = 0; i < K_NEI; ++i) {
    float d = yb[ib[i]] - xv;
    mx = d > mx ? d : mx;
  }
  __bf16* ob = outT + ((long long)b * N + n) * 2 * C;
  ob[2 * c]     = f2bf(xv);
  ob[2 * c + 1] = f2bf(mx);
}

__global__ void gpool_kernel(const float* __restrict__ in, float* __restrict__ out,
                             int C, int HW, long long total) {
  long long t = (long long)blockIdx.x * blockDim.x + threadIdx.x;
  if (t >= total) return;
  const float* ip = in + t * HW;
  float s = 0.0f;
  for (int i = 0; i < HW; ++i) s += ip[i];
  out[t] = s / (float)HW;
}

__global__ void head1_kernel(const float* __restrict__ pooled, const float* __restrict__ w,
                             const float* __restrict__ bias, const float* __restrict__ g,
                             const float* __restrict__ bb, float* __restrict__ out,
                             int Cin, int Co, long long total) {
  long long t = (long long)blockIdx.x * blockDim.x + threadIdx.x;
  if (t >= total) return;
  int o = (int)(t % Co); int b = (int)(t / Co);
  const float* pv = pooled + (long long)b * Cin;
  const float* wv = w + (long long)o * Cin;
  float s = 0.0f;
  for (int c = 0; c < Cin; ++c) s += pv[c] * wv[c];
  out[t] = gelu_f((s + bias[o]) * g[o] + bb[o]);
}

__global__ void head2_kernel(const float* __restrict__ h, const float* __restrict__ w,
                             const float* __restrict__ bias, float* __restrict__ out,
                             int Cin, int Co, long long total) {
  long long t = (long long)blockIdx.x * blockDim.x + threadIdx.x;
  if (t >= total) return;
  int o = (int)(t % Co); int b = (int)(t / Co);
  const float* hv = h + (long long)b * Cin;
  const float* wv = w + (long long)o * Cin;
  float s = 0.0f;
  for (int c = 0; c < Cin; ++c) s += hv[c] * wv[c];
  out[t] = s + bias[o];
}

// ---------------------------------------------------------------------------
struct GrapherP {
  const float *fc1_w, *fc1_b, *fc1_g, *fc1_bb;
  const float *fc2_w, *fc2_b, *fc2_g, *fc2_bb;
  const float *gc_w, *gc_b, *gc_g, *gc_bb;
  const float *rel;
};
struct FfnP {
  const float *fc1_w, *fc1_b, *g1, *b1;
  const float *dw_w, *dw_b, *g2, *b2;
  const float *fc2_w, *fc2_b, *g3, *b3;
};
struct DownP { const float *w, *b, *g, *bb; };

extern "C" void kernel_launch(void* const* d_in, const int* in_sizes, int n_in,
                              void* d_out, int out_size, void* d_ws, size_t ws_size,
                              hipStream_t stream) {
  (void)in_sizes; (void)n_in; (void)out_size; (void)ws_size;

  // param pointers: setup_inputs() dict insertion order, depth-first.
  int pi = 0;
  auto F = [&](void) -> const float* { return (const float*)d_in[pi++]; };

  const float* Xin = F();
  const float *sw1 = F(), *sb1 = F(), *sg1 = F(), *sbb1 = F();
  const float *sw2 = F(), *sb2 = F(), *sg2 = F(), *sbb2 = F();
  const float *sw3 = F(), *sb3 = F(), *sg3 = F(), *sbb3 = F();
  const float* pos = F();

  const int hasDown[12] = {0,0,1,0,1,0,0,0,0,0,1,0};
  GrapherP gp[12]; FfnP fp[12]; DownP dp[12];
  for (int i = 0; i < 12; ++i) {
    GrapherP& g = gp[i];
    g.fc1_w = F(); g.fc1_b = F(); g.fc1_g = F(); g.fc1_bb = F();
    g.fc2_w = F(); g.fc2_b = F(); g.fc2_g = F(); g.fc2_bb = F();
    g.gc_w  = F(); g.gc_b  = F(); g.gc_g  = F(); g.gc_bb  = F();
    g.rel   = F();
    FfnP& f = fp[i];
    f.fc1_w = F(); f.fc1_b = F(); f.g1 = F(); f.b1 = F();
    f.dw_w  = F(); f.dw_b  = F(); f.g2 = F(); f.b2 = F();
    f.fc2_w = F(); f.fc2_b = F(); f.g3 = F(); f.b3 = F();
    if (hasDown[i]) { dp[i].w = F(); dp[i].b = F(); dp[i].g = F(); dp[i].bb = F(); }
  }
  const float *pw1 = F(), *pb1 = F(), *pg = F(), *pbb = F(), *pw2 = F(), *pb2 = F();

  // workspace slabs
  char* base = (char*)d_ws;
  size_t off = 0;
  auto allocF = [&](size_t n) -> float* {
    float* p = (float*)(base + off); off += ((n * 4 + 255) & ~(size_t)255); return p;
  };
  auto allocH = [&](size_t n) -> __bf16* {
    __bf16* p = (__bf16*)(base + off); off += ((n * 2 + 255) & ~(size_t)255); return p;
  };
  float* ACT   = allocF(10485760);   // f32 residual stream [B,C,N]
  float* T1    = allocF(10485760);   // grapher fc1 out [B,C,N]
  float* Yb    = allocF(3276800);    // pooled candidates [B,C,M]
  float* SX    = allocF(131072);
  float* SY    = allocF(8192);
  float* TMPA  = allocF(41943040);   // dist matrix / ffn hidden (f32)
  float* TMPB  = allocF(10485760);   // down-conv temp / stem ping-pong
  float* XT    = allocF(20971520);   // transposed input
  float* POOL  = allocF(20480);
  float* H1    = allocF(32768);
  int*  IDXB   = (int*)allocF(1310720);
  __bf16* ACTH = allocH(10485760);   // [B,N,C] node-major
  __bf16* XNH  = allocH(10485760);   // [B,N,C]
  __bf16* YNH  = allocH(3276800);    // [B,M,C]
  __bf16* H2H  = allocH(20971520);   // [B,N,2C]
  __bf16* H3H  = allocH(20971520);   // [B,N,2C]
  __bf16* DWH  = allocH(41943040);   // [B,N,4C]
  __bf16* WH   = allocH(1703936);    // converted weights [M,K]

  const int TB = 256;
  auto grid1 = [&](long long n) { return dim3((unsigned)((n + TB - 1) / TB)); };

  auto gemm = [&](const float* Wf, const __bf16* BT, long long b_bs,
                  float* Cf, long long c_bs, __bf16* CbT, long long cb_bs,
                  const float* bias, const float* g, const float* bb,
                  const float* resid, int M, int N, int K, int act) {
    cvt_f32_bf16_kernel<<<grid1((long long)M * K), TB, 0, stream>>>(
        Wf, WH, (long long)M * K);
    dim3 gr(N / 64, (M + 63) / 64, BATCH);
    gemm_bf16_wmma_kernel<<<gr, 128, 0, stream>>>(
        WH, BT, b_bs, Cf, c_bs, CbT, cb_bs, bias, g, bb, resid, M, N, K, act);
  };
  auto cvtT = [&](const float* in, __bf16* outT, int C, int N) {
    long long tot = (long long)BATCH * C * N;
    cvtT_f32_bf16_kernel<<<grid1(tot), TB, 0, stream>>>(in, outT, C, N, tot);
  };

  // ---------------- stem ----------------
  transpose_in_kernel<<<grid1((long long)BATCH * 65536), TB, 0, stream>>>(
      Xin, XT, 1024, 64, (long long)BATCH * 65536);
  {
    long long tot = (long long)BATCH * 40 * 32 * 512;
    conv3x3_kernel<<<grid1(tot), TB, 0, stream>>>(XT, sw1, sb1, sg1, sbb1, TMPA,
        1, 40, 64, 1024, 32, 512, 2, 0, 0, 1, tot);
  }
  {
    long long tot = (long long)BATCH * 80 * 16 * 256;
    conv3x3_kernel<<<grid1(tot), TB, 0, stream>>>(TMPA, sw2, sb2, sg2, sbb2, TMPB,
        40, 80, 32, 512, 16, 256, 2, 0, 0, 1, tot);
  }
  {
    long long tot = (long long)BATCH * 80 * 16 * 256;
    conv3x3_kernel<<<grid1(tot), TB, 0, stream>>>(TMPB, sw3, sb3, sg3, sbb3, ACT,
        80, 80, 16, 256, 16, 256, 1, 1, 1, 0, tot);
    addpos_kernel<<<grid1(tot), TB, 0, stream>>>(ACT, pos, 80LL * 16 * 256, tot);
    cvtT(ACT, ACTH, 80, 4096);
  }

  // ---------------- blocks ----------------
  const int stageOf[12] = {0,0,1,1,2,2,2,2,2,2,3,3};
  const int Cs[4] = {80,160,400,640}, Hs[4] = {16,8,4,2}, Ws_[4] = {256,128,64,32};
  const int Rs[4] = {4,2,1,1};
  const int dil[12] = {1,1,1,1,2,2,2,2,3,3,3,3};
  int prevC = 80;

  for (int i = 0; i < 12; ++i) {
    const int st = stageOf[i];
    const int C = Cs[st], H = Hs[st], Wd = Ws_[st], rr = Rs[st];
    const int N = H * Wd;
    const int M = (rr > 1) ? N / (rr * rr) : N;
    const long long CN = (long long)C * N;

    if (hasDown[i]) {
      const int pH = Hs[st - 1], pW = Ws_[st - 1];
      long long tot = (long long)BATCH * C * N;
      conv3x3_kernel<<<grid1(tot), TB, 0, stream>>>(ACT, dp[i].w, dp[i].b, dp[i].g,
          dp[i].bb, TMPB, prevC, C, pH, pW, H, Wd, 2, 0, 0, 0, tot);
      hipMemcpyAsync(ACT, TMPB, (size_t)tot * sizeof(float),
                     hipMemcpyDeviceToDevice, stream);
      cvtT(ACT, ACTH, C, N);
    }

    // ---- grapher ----
    const GrapherP& g = gp[i];
    gemm(g.fc1_w, ACTH, CN, T1, CN, nullptr, 0, g.fc1_b, g.fc1_g, g.fc1_bb,
         nullptr, C, N, C, 0);

    const float* Yp; const __bf16* ynh; const float* syp;
    normalize_kernel<<<grid1((long long)BATCH * N), TB, 0, stream>>>(
        T1, XNH, SX, C, N, (long long)BATCH * N);
    if (rr > 1) {
      long long tot = (long long)BATCH * C * M;
      avgpool_kernel<<<grid1(tot), TB, 0, stream>>>(T1, Yb, C, H, Wd, rr, tot);
      normalize_kernel<<<grid1((long long)BATCH * M), TB, 0, stream>>>(
          Yb, YNH, SY, C, M, (long long)BATCH * M);
      Yp = Yb; ynh = YNH; syp = SY;
    } else {
      Yp = T1; ynh = XNH; syp = SX;     // candidates == queries
    }
    {
      dim3 gr(M / 64, N / 64, BATCH);
      dist_bf16_wmma_kernel<<<gr, 128, 0, stream>>>(XNH, ynh, SX, syp, g.rel,
                                                    TMPA, N, M, C);
    }
    topk_kernel<<<grid1((long long)BATCH * N), 128, 0, stream>>>(
        TMPA, IDXB, M, K_NEI * dil[i], dil[i], (long long)BATCH * N);
    {
      long long tot = (long long)BATCH * C * N;
      gather_maxrel_kernel<<<grid1(tot), TB, 0, stream>>>(T1, Yp, IDXB, H2H,
                                                          C, N, M, tot);
    }
    gemm(g.gc_w, H2H, 2 * CN, nullptr, 0, H3H, 2 * CN,
         g.gc_b, g.gc_g, g.gc_bb, nullptr, 2 * C, N, 2 * C, 1);
    gemm(g.fc2_w, H3H, 2 * CN, ACT, CN, ACTH, CN,
         g.fc2_b, g.fc2_g, g.fc2_bb, ACT, C, N, 2 * C, 0);

    // ---- ConvFFN ----
    const FfnP& f = fp[i];
    const int hid = 4 * C;
    const long long HN = (long long)hid * N;
    gemm(f.fc1_w, ACTH, CN, TMPA, HN, nullptr, 0, f.fc1_b, f.g1, f.b1,
         nullptr, hid, N, C, 1);
    {
      long long tot = (long long)BATCH * hid * N;
      dwconv3x3_kernel<<<grid1(tot), TB, 0, stream>>>(TMPA, f.dw_w, f.dw_b,
          f.g2, f.b2, DWH, hid, H, Wd, tot);
    }
    gemm(f.fc2_w, DWH, HN, ACT, CN, ACTH, CN, f.fc2_b, f.g3, f.b3,
         ACT, C, N, hid, 0);

    prevC = C;
  }

  // ---------------- head ----------------
  gpool_kernel<<<grid1((long long)BATCH * 640), TB, 0, stream>>>(
      ACT, POOL, 640, 64, (long long)BATCH * 640);
  head1_kernel<<<grid1((long long)BATCH * 1024), TB, 0, stream>>>(
      POOL, pw1, pb1, pg, pbb, H1, 640, 1024, (long long)BATCH * 1024);
  head2_kernel<<<grid1((long long)BATCH * 10), TB, 0, stream>>>(
      H1, pw2, pb2, (float*)d_out, 1024, 10, (long long)BATCH * 10);
}
